// SingleLevelTextGuidedCorrectionV2_85959475462326
// MI455X (gfx1250) — matrix operand order repaired and checked by reading, not compiled, not run on hardware
//
#include <hip/hip_runtime.h>

// ---------------------------------------------------------------------------
// CDNA5 (gfx1250): wave32 v_wmma_f32_16x16x32_bf16 GEMMs + TDM LDS staging.
// ---------------------------------------------------------------------------

typedef __bf16 bf16;
typedef __attribute__((ext_vector_type(2)))  __bf16 bf16x2;
typedef __attribute__((ext_vector_type(8)))  __bf16 bf16x8;
typedef __attribute__((ext_vector_type(16))) __bf16 bf16x16;
typedef __attribute__((ext_vector_type(8)))  float  f32x8;
typedef __attribute__((ext_vector_type(4)))  unsigned int uint32x4;
typedef __attribute__((ext_vector_type(8)))  int int32x8;
typedef __attribute__((ext_vector_type(4)))  int int32x4;

#define B_    8
#define C_    256
#define H_    128
#define W_    128
#define HW_   16384
#define NTOK_ 16

#if defined(__has_builtin)
# if __has_builtin(__builtin_amdgcn_tensor_load_to_lds) && \
     __has_builtin(__builtin_amdgcn_s_wait_tensorcnt)
#  define HAVE_TDM 1
# endif
#endif
#ifndef HAVE_TDM
# define HAVE_TDM 0
#endif

union AFrag { bf16x16 v; bf16 e[16]; };
union Acc   { f32x8   v; float e[8]; };
union Pk8   { bf16x8  v; bf16 e[8]; };

__device__ __forceinline__ int laneid() { return threadIdx.x & 31; }

__device__ __forceinline__ void zero_acc(Acc& a) {
#pragma unroll
  for (int r = 0; r < 8; r++) a.e[r] = 0.f;
}

__device__ __forceinline__ void store_pk(bf16* p, float a, float b) {
  bf16x2 v; v.x = (bf16)a; v.y = (bf16)b;
  *(bf16x2*)p = v;
}

__device__ __forceinline__ f32x8 wmma_bf16(bf16x16 a, bf16x16 b, f32x8 c) {
  return __builtin_amdgcn_wmma_f32_16x16x32_bf16(false, a, false, b,
                                                 (short)0, c, false, false);
}

// A fragment (16x32 bf16) from LDS [m][k], stride mult of 8.
// lane: m = lane&15, half = lane>>4; element i -> k = (i<8 ? i : i+8) + 8*half.
__device__ __forceinline__ bf16x16 frag_a_lds(const bf16* base, int stride) {
  const int l = laneid();
  const bf16* p = base + (l & 15) * stride + ((l >> 4) << 3);
  bf16x8 lo = *(const bf16x8*)(p);
  bf16x8 hi = *(const bf16x8*)(p + 16);
  return __builtin_shufflevector(lo, hi, 0,1,2,3,4,5,6,7,8,9,10,11,12,13,14,15);
}

// B fragment (32x16 bf16) from LDS [n][k], stride mult of 8.
// lane: n = lane&15, half = lane>>4; element i -> k = (i<8 ? i : i+8) + 8*half.
__device__ __forceinline__ bf16x16 frag_b_lds(const bf16* base, int stride) {
  const int l = laneid();
  const bf16* p = base + (l & 15) * stride + ((l >> 4) << 3);
  bf16x8 lo = *(const bf16x8*)(p);
  bf16x8 hi = *(const bf16x8*)(p + 16);
  return __builtin_shufflevector(lo, hi, 0,1,2,3,4,5,6,7,8,9,10,11,12,13,14,15);
}

#if HAVE_TDM
// Issue a 2D TDM tile load: tile_d0 contiguous bf16 elements per row,
// tile_d1 rows with row_stride (elements) in global; LDS rows get
// (pad_amount+1) DWORDs of padding after each 2^(pad_interval+3) bytes.
__device__ __forceinline__ void tdm_load_2d(unsigned lds_byte, const void* gaddr,
                                            int tile_d0, int tile_d1,
                                            long long row_stride_elems,
                                            int pad_interval_code,
                                            int pad_amount_code) {
  unsigned long long ga = (unsigned long long)(uintptr_t)gaddr;
  uint32x4 g0;
  g0[0] = 1u;                                        // count=1, user mode
  g0[1] = lds_byte;                                  // lds_addr
  g0[2] = (unsigned)(ga & 0xffffffffull);            // global_addr[31:0]
  g0[3] = (unsigned)((ga >> 32) & 0x01ffffffull) | (2u << 30);  // type=2
  const unsigned td0 = 1u << 20, td1 = 1u << 20;     // large dims (no OOB)
  unsigned long long s0 = (unsigned long long)row_stride_elems;
  int32x8 g1;
  g1[0] = (int)((1u << 16) |                         // data_size = 2 bytes
                (1u << 20) |                         // pad_enable
                ((unsigned)pad_interval_code << 22) |
                ((unsigned)pad_amount_code << 25));
  g1[1] = (int)((td0 & 0xffffu) << 16);              // tensor_dim0 lo16
  g1[2] = (int)(((td0 >> 16) & 0xffffu) | ((td1 & 0xffffu) << 16));
  g1[3] = (int)(((td1 >> 16) & 0xffffu) | ((unsigned)tile_d0 << 16));
  g1[4] = (int)((unsigned)tile_d1 & 0xffffu);        // tile_dim1; tile_dim2=0
  g1[5] = (int)(s0 & 0xffffffffull);                 // dim0 stride lo32
  g1[6] = (int)((s0 >> 32) & 0xffffull);             // dim0 stride hi16
  g1[7] = 0;
  int32x4 z4 = {};
  int32x8 z8 = {};
  __builtin_amdgcn_tensor_load_to_lds(g0, g1, z4, z4, z8, 0);
}
#endif

// ---------------------------------------------------------------------------
// K1a: Q[b,n,d] = txt[b,n,:] . wq[d,:] + bq[d]
// ---------------------------------------------------------------------------
__global__ void qproj_kernel(const float* __restrict__ txt,
                             const float* __restrict__ wq,
                             const float* __restrict__ bq,
                             float* __restrict__ Q) {
  int idx = blockIdx.x * 256 + threadIdx.x;
  int d = idx & 127, bn = idx >> 7;
  const float* t = txt + (size_t)bn * 512;
  const float* w = wq + (size_t)d * 512;
  float s = bq[d];
  for (int k = 0; k < 512; k++) s += t[k] * w[k];
  Q[idx] = s;
}

// K1b: Qw[mod,b,n,c] = sum_d Q[b,n,d] * wk[d,c]
__global__ void qw_kernel(const float* __restrict__ Q,
                          const float* __restrict__ wk_rgb,
                          const float* __restrict__ wk_ir,
                          float* __restrict__ Qw) {
  int idx = blockIdx.x * 256 + threadIdx.x;
  int c = idx & 255, bn = (idx >> 8) & 127, mod = idx >> 15;
  const float* wk = mod ? wk_ir : wk_rgb;
  const float* q = Q + (size_t)bn * 128;
  float s = 0.f;
  for (int d = 0; d < 128; d++) s += q[d] * wk[d * 256 + c];
  Qw[idx] = s;
}

// K1c: qb[mod,b,n] = sum_d Q[b,n,d] * bk[d]
__global__ void qb_kernel(const float* __restrict__ Q,
                          const float* __restrict__ bk_rgb,
                          const float* __restrict__ bk_ir,
                          float* __restrict__ qb) {
  int idx = threadIdx.x;
  int bn = idx & 127, mod = idx >> 7;
  const float* bk = mod ? bk_ir : bk_rgb;
  float s = 0.f;
  for (int d = 0; d < 128; d++) s += Q[bn * 128 + d] * bk[d];
  qb[idx] = s;
}

// ---------------------------------------------------------------------------
// K2: logits S = (Qw @ X + qb) / sqrt(128).  grid (HW/64, B, 2), 128 thr.
// ---------------------------------------------------------------------------
__launch_bounds__(128)
__global__ void logits_wmma(const float* __restrict__ Qw,
                            const float* __restrict__ qb,
                            const float* __restrict__ xrgb,
                            const float* __restrict__ xir,
                            float* __restrict__ Ab) {
  __shared__ __attribute__((aligned(16))) bf16 sA[16 * 40];
  __shared__ __attribute__((aligned(16))) bf16 sB[64 * 40];
  const int tid = threadIdx.x, lane = tid & 31, wv = tid >> 5;
  const int b = blockIdx.y, mod = blockIdx.z;
  const int hw0 = blockIdx.x * 64;
  const float* X = (mod == 0) ? xrgb : xir;
  const float* Qwb = Qw + (size_t)(mod * B_ + b) * NTOK_ * C_;
  const float* qbb = qb + (size_t)(mod * B_ + b) * NTOK_;
  const size_t xbase = (size_t)b * C_ * HW_;
  const int kA2 = (tid & 15) << 1, mA = tid >> 4;   // mA 0..7
  const int nld = tid & 63, kB2 = (tid >> 6) << 1;  // kB2 0,2

  Acc acc; zero_acc(acc);
  for (int kk = 0; kk < C_; kk += 32) {
    __syncthreads();
#pragma unroll
    for (int i = 0; i < 2; i++) {
      int m = mA + i * 8;
      const float* wp = Qwb + (size_t)m * C_ + kk + kA2;
      store_pk(&sA[m * 40 + kA2], wp[0], wp[1]);
    }
#pragma unroll
    for (int i = 0; i < 8; i++) {
      int k = kB2 + i * 4;
      float v0 = X[xbase + (size_t)(kk + k) * HW_ + hw0 + nld];
      float v1 = X[xbase + (size_t)(kk + k + 1) * HW_ + hw0 + nld];
      store_pk(&sB[nld * 40 + k], v0, v1);
    }
    __syncthreads();
    bf16x16 af = frag_a_lds(sA, 40);
    bf16x16 bf_ = frag_b_lds(sB + wv * 16 * 40, 40);
    acc.v = wmma_bf16(af, bf_, acc.v);
  }
  const int ncol = lane & 15, mh8 = (lane >> 4) << 3;
  const float scale = 0.088388347648318447f;
  int hw = hw0 + wv * 16 + ncol;
#pragma unroll
  for (int r = 0; r < 8; r++) {
    int nr = r + mh8;
    Ab[((size_t)(mod * B_ + b) * NTOK_ + nr) * HW_ + hw] =
        (acc.e[r] + qbb[nr]) * scale;
  }
}

// ---------------------------------------------------------------------------
// K3: in-place softmax along HW for each of the 256 rows.
// ---------------------------------------------------------------------------
__global__ void softmax_kernel(float* __restrict__ A) {
  const int tid = threadIdx.x;
  float* p = A + (size_t)blockIdx.x * HW_;
  __shared__ float red[256];
  float m = -3.4e38f;
  for (int i = tid; i < HW_; i += 256) m = fmaxf(m, p[i]);
  red[tid] = m; __syncthreads();
  for (int s = 128; s > 0; s >>= 1) {
    if (tid < s) red[tid] = fmaxf(red[tid], red[tid + s]);
    __syncthreads();
  }
  m = red[0]; __syncthreads();
  float sum = 0.f;
  for (int i = tid; i < HW_; i += 256) {
    float e = __expf(p[i] - m);
    p[i] = e; sum += e;
  }
  red[tid] = sum; __syncthreads();
  for (int s = 128; s > 0; s >>= 1) {
    if (tid < s) red[tid] += red[tid + s];
    __syncthreads();
  }
  float r = 1.f / red[0];
  for (int i = tid; i < HW_; i += 256) p[i] *= r;
}

// K4: G[b,n] = clip(<A_rgb,A_ir>/(||A_rgb|| ||A_ir||), 0, 1)
__global__ void gcoh_kernel(const float* __restrict__ A, float* __restrict__ G) {
  const int tid = threadIdx.x, bn = blockIdx.x;
  const float* ar = A + (size_t)bn * HW_;
  const float* ai = A + (size_t)(128 + bn) * HW_;
  __shared__ float red[3][256];
  float d = 0.f, s1 = 0.f, s2 = 0.f;
  for (int i = tid; i < HW_; i += 256) {
    float x = ar[i], y = ai[i];
    d += x * y; s1 += x * x; s2 += y * y;
  }
  red[0][tid] = d; red[1][tid] = s1; red[2][tid] = s2; __syncthreads();
  for (int s = 128; s > 0; s >>= 1) {
    if (tid < s) {
      red[0][tid] += red[0][tid + s];
      red[1][tid] += red[1][tid + s];
      red[2][tid] += red[2][tid + s];
    }
    __syncthreads();
  }
  if (tid == 0) {
    float nr = fmaxf(sqrtf(red[1][0]), 1e-12f);
    float ni = fmaxf(sqrtf(red[2][0]), 1e-12f);
    G[bn] = fminf(fmaxf(red[0][0] / (nr * ni), 0.f), 1.f);
  }
}

// K5: M_err[b,hw] = sum_n (1-G[b,n]) * |A_rgb - A_ir|
__global__ void merr_kernel(const float* __restrict__ A,
                            const float* __restrict__ G,
                            float* __restrict__ M) {
  const int b = blockIdx.y;
  const int hw = blockIdx.x * 256 + threadIdx.x;
  float acc = 0.f;
#pragma unroll
  for (int n = 0; n < 16; n++) {
    int bn = b * 16 + n;
    float x = A[(size_t)bn * HW_ + hw];
    float y = A[(size_t)(bn + 128) * HW_ + hw];
    acc += (1.f - G[bn]) * fabsf(x - y);
  }
  M[(size_t)b * HW_ + hw] = acc;
}

// K6: per-batch min/max of M_err
__global__ void minmax_kernel(const float* __restrict__ M,
                              float* __restrict__ mn, float* __restrict__ mx) {
  const int tid = threadIdx.x, b = blockIdx.x;
  __shared__ float rlo[256], rhi[256];
  float lo = 3.4e38f, hi = -3.4e38f;
  for (int i = tid; i < HW_; i += 256) {
    float v = M[(size_t)b * HW_ + i];
    lo = fminf(lo, v); hi = fmaxf(hi, v);
  }
  rlo[tid] = lo; rhi[tid] = hi; __syncthreads();
  for (int s = 128; s > 0; s >>= 1) {
    if (tid < s) {
      rlo[tid] = fminf(rlo[tid], rlo[tid + s]);
      rhi[tid] = fmaxf(rhi[tid], rhi[tid + s]);
    }
    __syncthreads();
  }
  if (tid == 0) { mn[b] = rlo[0]; mx[b] = rhi[0]; }
}

// ---------------------------------------------------------------------------
// conv1x1 WMMA: Out[256, 64 hw] = W[256,256] @ X[256, 64].
// MODE 0: X = x_ir([c][hw] f32) * Mnorm, epilogue BN1+ReLU -> h1 ([hw][c] bf16)
// MODE 1: X = h2 ([hw][c] bf16, TDM-staged), epilogue x_ir - alpha*(acc+b)
// grid (HW/64, B), 256 threads (8 waves); wave: 2 m-tiles x 4 n-tiles.
// ---------------------------------------------------------------------------
template <int MODE>
__launch_bounds__(256)
__global__ void conv1x1_wmma(const float* __restrict__ Wt,
                             const float* __restrict__ Xf,
                             const bf16* __restrict__ Xb,
                             const float* __restrict__ Me,
                             const float* __restrict__ mnv,
                             const float* __restrict__ mxv,
                             const float* __restrict__ bng,
                             const float* __restrict__ bnb,
                             const float* __restrict__ bnm,
                             const float* __restrict__ bnv,
                             const float* __restrict__ bias,
                             const float* __restrict__ xres,
                             const float* __restrict__ alpha,
                             bf16* __restrict__ outb,
                             float* __restrict__ outf) {
  __shared__ __attribute__((aligned(16))) bf16 sA[C_ * 40];
  __shared__ __attribute__((aligned(16))) bf16 sB[64 * 40];
  const int tid = threadIdx.x, lane = tid & 31, wv = tid >> 5;
  const int b = blockIdx.y;
  const int hw0 = blockIdx.x * 64;
  const size_t xbase = (size_t)b * C_ * HW_;
  const int kA2 = (tid & 15) << 1, mA = tid >> 4;   // mA 0..15
  const int nld = tid & 63, kB2 = (tid >> 6) << 1;  // kB2 0,2,4,6
  const int k8 = (tid >> 6) << 3;                   // 0,8,16,24

  float mval = 0.f;
  if (MODE == 0) {
    float mmn = mnv[b];
    float msc = 1.f / fmaxf(mxv[b] - mmn, 1e-6f);
    mval = (Me[(size_t)b * HW_ + hw0 + nld] - mmn) * msc;
  }

  Acc acc[2][4];
#pragma unroll
  for (int mi = 0; mi < 2; mi++)
#pragma unroll
    for (int ni = 0; ni < 4; ni++) zero_acc(acc[mi][ni]);

  for (int kk = 0; kk < C_; kk += 32) {
    __syncthreads();
#pragma unroll
    for (int i = 0; i < 16; i++) {            // stage W[:, kk:kk+32] as bf16
      int m = mA + i * 16;
      const float* wp = Wt + (size_t)m * C_ + kk + kA2;
      store_pk(&sA[m * 40 + kA2], wp[0], wp[1]);
    }
    if (MODE == 0) {                          // stage x_ir * mask ([c][hw] f32)
#pragma unroll
      for (int i = 0; i < 4; i++) {
        int k = kB2 + i * 8;
        float v0 = Xf[xbase + (size_t)(kk + k) * HW_ + hw0 + nld] * mval;
        float v1 = Xf[xbase + (size_t)(kk + k + 1) * HW_ + hw0 + nld] * mval;
        store_pk(&sB[nld * 40 + k], v0, v1);
      }
    } else {                                  // stage h2 ([hw][c] bf16)
#if HAVE_TDM
      if (wv == 0) {
        tdm_load_2d((unsigned)(uintptr_t)sB,
                    Xb + ((size_t)b * HW_ + hw0) * C_ + kk,
                    /*tile_d0=*/32, /*tile_d1=*/64, /*row_stride=*/C_,
                    /*pad_interval(64B)=*/3, /*pad_amount(16B)=*/3);
        __builtin_amdgcn_s_wait_tensorcnt(0);
      }
#else
      bf16x8 v = *(const bf16x8*)(Xb + ((size_t)b * HW_ + hw0 + nld) * C_ + kk + k8);
      *(bf16x8*)&sB[nld * 40 + k8] = v;
#endif
    }
    __syncthreads();
#pragma unroll
    for (int mi = 0; mi < 2; mi++) {
      bf16x16 af = frag_a_lds(sA + (wv * 2 + mi) * 16 * 40, 40);
#pragma unroll
      for (int ni = 0; ni < 4; ni++) {
        bf16x16 bf_ = frag_b_lds(sB + ni * 16 * 40, 40);
        acc[mi][ni].v = wmma_bf16(af, bf_, acc[mi][ni].v);
      }
    }
  }

  const int ncol = lane & 15, mh8 = (lane >> 4) << 3;
  float al = (MODE == 1) ? alpha[0] : 0.f;
#pragma unroll
  for (int mi = 0; mi < 2; mi++) {
    int mt = (wv * 2 + mi) * 16;
#pragma unroll
    for (int ni = 0; ni < 4; ni++) {
      int hw = hw0 + ni * 16 + ncol;
      if (MODE == 0) {                         // h1 [hw][c], packed 16B store
        Pk8 o;
#pragma unroll
        for (int r = 0; r < 8; r++) {
          int ch = mt + mh8 + r;
          float inv = bng[ch] * rsqrtf(bnv[ch] + 1e-5f);
          float v = fmaxf(acc[mi][ni].e[r] * inv + (bnb[ch] - bnm[ch] * inv), 0.f);
          o.e[r] = (bf16)v;
        }
        *(bf16x8*)&outb[((size_t)(b * HW_ + hw)) * C_ + mt + mh8] = o.v;
      } else {                                 // final residual, f32 [c][hw]
#pragma unroll
        for (int r = 0; r < 8; r++) {
          int ch = mt + mh8 + r;
          size_t o = xbase + (size_t)ch * HW_ + hw;
          outf[o] = xres[o] - al * (acc[mi][ni].e[r] + bias[ch]);
        }
      }
    }
  }
}

// ---------------------------------------------------------------------------
// conv3x3 WMMA (implicit GEMM, 9 taps). grid (H, B, 2), 256 threads.
// h1 is [b][y][x][c] bf16 (channel-contiguous) -> TDM stages 64-channel
// chunks of rows y-1..y+1 into LDS [plane][x][c], stride 72 via pad_amount.
// ---------------------------------------------------------------------------
__launch_bounds__(256)
__global__ void conv3x3_wmma(const bf16* __restrict__ h1,
                             const float* __restrict__ w2,
                             const float* __restrict__ bng,
                             const float* __restrict__ bnb,
                             const float* __restrict__ bnm,
                             const float* __restrict__ bnv,
                             bf16* __restrict__ h2) {
  const int CCP = 72, PS = 130 * 72;
  __shared__ __attribute__((aligned(16))) bf16 sX[3 * 130 * 72];  // 56,160 B
  const int tid = threadIdx.x, lane = tid & 31, wv = tid >> 5;
  const int y = blockIdx.x, b = blockIdx.y, mh = blockIdx.z;
  const int m0 = mh * 128 + wv * 16;

  Acc acc[8];
#pragma unroll
  for (int nt = 0; nt < 8; nt++) zero_acc(acc[nt]);

  for (int kc = 0; kc < C_; kc += 64) {
    __syncthreads();
    // zero the x = -1 and x = 128 pad columns
    for (int i = tid; i < 384; i += 256) {
      int p = i >> 7, r = i & 127;
      int col = (r >= 64) ? 129 : 0, c = r & 63;
      sX[p * PS + col * CCP + c] = (bf16)0.f;
    }
#if HAVE_TDM
    // zero out-of-range planes (image top/bottom rows)
#pragma unroll
    for (int p = 0; p < 3; p++) {
      int yy = y + p - 1;
      if (yy < 0 || yy >= H_) {
        for (int i = tid; i < 1024; i += 256) {
          int x = i >> 3, c8 = (i & 7) << 3;
          bf16x8 z = {};
          *(bf16x8*)&sX[p * PS + (x + 1) * CCP + c8] = z;
        }
      }
    }
    if (wv == 0) {
#pragma unroll
      for (int p = 0; p < 3; p++) {
        int yy = y + p - 1;
        if (yy >= 0 && yy < H_) {
          const bf16* g = h1 + ((size_t)(b * HW_ + yy * W_)) * C_ + kc;
          tdm_load_2d((unsigned)(uintptr_t)sX + (unsigned)((p * PS + CCP) * 2),
                      g, /*tile_d0=*/64, /*tile_d1=*/128, /*row_stride=*/C_,
                      /*pad_interval(128B)=*/4, /*pad_amount(16B)=*/3);
        }
      }
      __builtin_amdgcn_s_wait_tensorcnt(0);
    }
#else
    for (int i = 0; i < 12; i++) {
      int g = tid + (i << 8);                 // 0..3071 16B chunks
      int c8 = (g & 7) << 3;
      int row = g >> 3;                       // 0..383
      int x = row & 127, p = row >> 7;
      int yy = y + p - 1;
      bf16x8 v = {};
      if (yy >= 0 && yy < H_)
        v = *(const bf16x8*)(h1 + ((size_t)(b * HW_ + yy * W_ + x)) * C_ + kc + c8);
      *(bf16x8*)&sX[p * PS + (x + 1) * CCP + c8] = v;
    }
#endif
    __syncthreads();
#pragma unroll
    for (int tap = 0; tap < 9; tap++) {
      const int dy = tap / 3, dx = tap % 3 - 1;
#pragma unroll
      for (int ks = 0; ks < 2; ks++) {
        AFrag af;
        const int mrow = m0 + (lane & 15);
        const int h8 = (lane >> 4) << 3;
#pragma unroll
        for (int i = 0; i < 16; i++) {
          int k = ((i < 8) ? i : (i + 8)) + h8 + ks * 32;
          af.e[i] = (bf16)w2[((size_t)mrow * C_ + kc + k) * 9 + tap];
        }
#pragma unroll
        for (int nt = 0; nt < 8; nt++) {
          const bf16* pB = sX + dy * PS +
                           (nt * 16 + (lane & 15) + dx + 1) * CCP +
                           ks * 32 + ((lane >> 4) << 3);
          bf16x8 lo = *(const bf16x8*)pB;
          bf16x8 hi = *(const bf16x8*)(pB + 16);
          bf16x16 bf_ = __builtin_shufflevector(
              lo, hi, 0,1,2,3,4,5,6,7,8,9,10,11,12,13,14,15);
          acc[nt].v = wmma_bf16(af.v, bf_, acc[nt].v);
        }
      }
    }
  }
  // BN2 + ReLU epilogue -> h2 [hw][c], packed 16B stores
  const int ncol = lane & 15, mh8 = (lane >> 4) << 3;
#pragma unroll
  for (int nt = 0; nt < 8; nt++) {
    int x = nt * 16 + ncol;
    Pk8 o;
#pragma unroll
    for (int r = 0; r < 8; r++) {
      int ch = m0 + mh8 + r;
      float inv = bng[ch] * rsqrtf(bnv[ch] + 1e-5f);
      o.e[r] = (bf16)fmaxf(acc[nt].e[r] * inv + (bnb[ch] - bnm[ch] * inv), 0.f);
    }
    *(bf16x8*)&h2[((size_t)(b * HW_ + y * W_ + x)) * C_ + m0 + mh8] = o.v;
  }
}

// ---------------------------------------------------------------------------
extern "C" void kernel_launch(void* const* d_in, const int* in_sizes, int n_in,
                              void* d_out, int out_size, void* d_ws,
                              size_t ws_size, hipStream_t stream) {
  const float* x_rgb  = (const float*)d_in[0];
  const float* x_ir   = (const float*)d_in[1];
  const float* txt    = (const float*)d_in[2];
  const float* alpha  = (const float*)d_in[3];
  const float* wq     = (const float*)d_in[4];
  const float* bq     = (const float*)d_in[5];
  const float* w_krgb = (const float*)d_in[6];
  const float* b_krgb = (const float*)d_in[7];
  const float* w_kir  = (const float*)d_in[8];
  const float* b_kir  = (const float*)d_in[9];
  const float* c1_w   = (const float*)d_in[10];
  const float* bn1_g  = (const float*)d_in[11];
  const float* bn1_b  = (const float*)d_in[12];
  const float* bn1_m  = (const float*)d_in[13];
  const float* bn1_v  = (const float*)d_in[14];
  const float* c2_w   = (const float*)d_in[15];
  const float* bn2_g  = (const float*)d_in[16];
  const float* bn2_b  = (const float*)d_in[17];
  const float* bn2_m  = (const float*)d_in[18];
  const float* bn2_v  = (const float*)d_in[19];
  const float* c3_w   = (const float*)d_in[20];
  const float* c3_b   = (const float*)d_in[21];

  char* ws = (char*)d_ws;
  size_t off = 0;
  auto take = [&](size_t bytes) -> char* {
    char* p = ws + off;
    off = (off + bytes + 255) & ~(size_t)255;
    return p;
  };
  float* Q  = (float*)take((size_t)B_ * NTOK_ * 128 * 4);
  float* Qw = (float*)take((size_t)2 * B_ * NTOK_ * C_ * 4);
  float* qb = (float*)take((size_t)2 * B_ * NTOK_ * 4);
  float* Gv = (float*)take((size_t)B_ * NTOK_ * 4);
  float* mn = (float*)take((size_t)B_ * 4);
  float* mx = (float*)take((size_t)B_ * 4);
  float* Ab = (float*)take((size_t)2 * B_ * NTOK_ * HW_ * 4);
  float* Me = (float*)take((size_t)B_ * HW_ * 4);
  bf16*  h1 = (bf16*)take((size_t)B_ * HW_ * C_ * 2);   // [b][hw][c]
  bf16*  h2 = (bf16*)take((size_t)B_ * HW_ * C_ * 2);   // [b][hw][c]
  float* out = (float*)d_out;

  qproj_kernel<<<64, 256, 0, stream>>>(txt, wq, bq, Q);
  qw_kernel<<<256, 256, 0, stream>>>(Q, w_krgb, w_kir, Qw);
  qb_kernel<<<1, 256, 0, stream>>>(Q, b_krgb, b_kir, qb);
  logits_wmma<<<dim3(HW_ / 64, B_, 2), 128, 0, stream>>>(Qw, qb, x_rgb, x_ir, Ab);
  softmax_kernel<<<2 * B_ * NTOK_, 256, 0, stream>>>(Ab);
  gcoh_kernel<<<B_ * NTOK_, 256, 0, stream>>>(Ab, Gv);
  merr_kernel<<<dim3(HW_ / 256, B_), 256, 0, stream>>>(Ab, Gv, Me);
  minmax_kernel<<<B_, 256, 0, stream>>>(Me, mn, mx);
  conv1x1_wmma<0><<<dim3(HW_ / 64, B_), 256, 0, stream>>>(
      c1_w, x_ir, nullptr, Me, mn, mx, bn1_g, bn1_b, bn1_m, bn1_v,
      nullptr, nullptr, nullptr, h1, nullptr);
  conv3x3_wmma<<<dim3(H_, B_, 2), 256, 0, stream>>>(
      h1, c2_w, bn2_g, bn2_b, bn2_m, bn2_v, h2);
  conv1x1_wmma<1><<<dim3(HW_ / 64, B_), 256, 0, stream>>>(
      c3_w, nullptr, h2, nullptr, nullptr, nullptr, nullptr, nullptr,
      nullptr, nullptr, c3_b, x_ir, alpha, nullptr, out);
  (void)in_sizes; (void)n_in; (void)out_size; (void)ws_size;
}
